// DynamicSTGTransformer_1185410974012
// MI455X (gfx1250) — compile-verified, matrix operand order checked by reading, not run on hardware
//
#include <hip/hip_runtime.h>
#include <math.h>

#define B_    2
#define FEAT_ 4
#define V_    50
#define T_    24
#define TT_   48
#define D_    256
#define H_    8
#define L_    6
#define N_    2400
#define DH_   32   // D/H

typedef __attribute__((ext_vector_type(16))) _Float16 v16h;
typedef __attribute__((ext_vector_type(8)))  _Float16 v8h;
typedef __attribute__((ext_vector_type(8)))  float    v8f;

// ---------------------------------------------------------------------------
// Fragment loader: A/B operand for v_wmma_f32_16x16x32_f16 (from global).
// Per CDNA5 ISA 16-bit A 16x32 layout:
//   lanes 0-15  : row M=lane,    elems 0..7 = K[k0..k0+7],   8..15 = K[k0+16..k0+23]
//   lanes 16-31 : row M=lane-16, elems 0..7 = K[k0+8..k0+15],8..15 = K[k0+24..k0+31]
// B fragment is identical with "row" = output column (K-contiguous column of W,
// i.e. a row of the pre-transposed weight).
// ---------------------------------------------------------------------------
__device__ __forceinline__ v16h load_frag16(const _Float16* base, int ld,
                                            int row0, int k0, int lane) {
  int r    = row0 + (lane & 15);
  int koff = (lane >> 4) << 3;
  const _Float16* p = base + (size_t)r * ld + k0 + koff;
  v8h lo = *(const v8h*)p;         // global_load_b128
  v8h hi = *(const v8h*)(p + 16);  // global_load_b128
  v16h f;
#pragma unroll
  for (int i = 0; i < 8; ++i) { f[i] = lo[i]; f[i + 8] = hi[i]; }
  return f;
}

__device__ __forceinline__ float gelu_f(float x) {
  return 0.5f * x * (1.0f + erff(x * 0.70710678118654752f));
}
__device__ __forceinline__ float silu_f(float x) {
  return x / (1.0f + __expf(-x));
}

// ---------------------------------------------------------------------------
// Generic WMMA GEMM:  out[M,Nc] = A[M,K](f16) @ WT[Nc,K](f16)^T + bias
// Block = 8 waves sharing one 16-row A panel:
//   * A panel (16 x K f16) staged global->LDS with GLOBAL_LOAD_ASYNC_TO_LDS_B128
//     (ASYNCcnt path), synced with s_wait_asynccnt + barrier.
//   * Each wave owns one 16-col tile (tn = group*8 + wave) and streams its
//     WT panel from global (hits the 192MB L2), A fragments via ds_load_b128.
// Epilogue: +bias, optional exact GELU, optional fp32 residual, f32/f16 stores.
// ---------------------------------------------------------------------------
__global__ void gemm_wmma_f16(const _Float16* __restrict__ A,
                              const _Float16* __restrict__ WT,
                              const float* __restrict__ bias,
                              const float* resid, float* outf,
                              _Float16* outh, int Nc, int K, int dogelu,
                              int tnGroups) {
  extern __shared__ _Float16 Abuf[];          // 16 * K halves
  int g  = blockIdx.x % tnGroups;
  int tm = blockIdx.x / tnGroups;
  int wave = threadIdx.x >> 5, lane = threadIdx.x & 31;
  int tn = g * 8 + wave;

  // ---- async-stage the A panel: 16 rows x K halves = 2*K 16-byte chunks ----
  int chunksPerRow = K >> 3;                  // 16B chunks per row
  for (int c = threadIdx.x; c < 2 * K; c += 256) {
    int row = c / chunksPerRow;
    int cc  = c - row * chunksPerRow;
    const _Float16* src = A + (size_t)(tm * 16 + row) * K + cc * 8;
    unsigned ldsoff = (unsigned)(unsigned long long)&Abuf[row * K + cc * 8];
    asm volatile("global_load_async_to_lds_b128 %0, %1, off"
                 :: "v"(ldsoff), "v"(src) : "memory");
  }
  asm volatile("s_wait_asynccnt 0x0" ::: "memory");
  __syncthreads();

  v8f acc = {};
  int rA    = lane & 15;
  int koffA = (lane >> 4) << 3;
  for (int k0 = 0; k0 < K; k0 += 32) {
    // A fragment from LDS (ds_load_b128 x2)
    const _Float16* pa = &Abuf[rA * K + k0 + koffA];
    v8h lo = *(const v8h*)pa;
    v8h hi = *(const v8h*)(pa + 16);
    v16h a;
#pragma unroll
    for (int i = 0; i < 8; ++i) { a[i] = lo[i]; a[i + 8] = hi[i]; }
    // B fragment from global/L2
    v16h b = load_frag16(WT, K, tn * 16, k0, lane);
    acc = __builtin_amdgcn_wmma_f32_16x16x32_f16(false, a, false, b,
                                                 (short)0, acc, false, false);
  }
  // C layout: VGPR r -> M = r + 8*(lane>=16), N = lane&15
  int row0 = tm * 16 + ((lane >> 4) << 3);
  int col  = tn * 16 + (lane & 15);
  float bv = bias ? bias[col] : 0.0f;
#pragma unroll
  for (int r = 0; r < 8; ++r) {
    size_t idx = (size_t)(row0 + r) * Nc + col;
    float v = acc[r] + bv;
    if (dogelu) v = gelu_f(v);
    if (resid)  v += resid[idx];
    if (outf) outf[idx] = v;
    if (outh) outh[idx] = (_Float16)v;
  }
}

// ---------------------------------------------------------------------------
// Weight transpose + f32->f16 convert: src (count, K, Nc) -> dst (count, Nc, K)
// ---------------------------------------------------------------------------
__global__ void wtrans_kernel(const float* __restrict__ src,
                              _Float16* __restrict__ dst,
                              int K, int Nc, int count) {
  size_t per = (size_t)K * Nc;
  size_t i = (size_t)blockIdx.x * blockDim.x + threadIdx.x;
  if (i >= per * count) return;
  int c = (int)(i / per);
  size_t r = i - (size_t)c * per;
  int n = (int)(r / K);
  int k = (int)(r - (size_t)n * K);
  dst[(size_t)c * per + (size_t)n * K + k] =
      (_Float16)src[(size_t)c * per + (size_t)k * Nc + n];
}

// ---------------------------------------------------------------------------
// h[b,n,d] = sum_f xc[b,f,v,tt]*xproj_w[f,d] + xproj_b[d]
//            + sp_emb[n%V][d] + tp_emb[n%TT][d]
// ---------------------------------------------------------------------------
__global__ void embed_kernel(const float* __restrict__ x,
                             const float* __restrict__ xm,
                             const float* __restrict__ w,
                             const float* __restrict__ bias,
                             const float* __restrict__ sp,
                             const float* __restrict__ tp,
                             float* __restrict__ h) {
  size_t i = (size_t)blockIdx.x * blockDim.x + threadIdx.x;
  if (i >= (size_t)B_ * N_ * D_) return;
  int d = (int)(i % D_);
  int n = (int)((i / D_) % N_);
  int b = (int)(i / ((size_t)D_ * N_));
  int v = n / TT_, tt = n - v * TT_;
  float acc = bias[d];
#pragma unroll
  for (int f = 0; f < FEAT_; ++f) {
    float xv;
    if (tt < T_) xv = x [(((size_t)b * FEAT_ + f) * V_ + v) * T_ + tt];
    else         xv = xm[(((size_t)b * FEAT_ + f) * V_ + v) * T_ + (tt - T_)];
    acc += xv * w[f * D_ + d];
  }
  acc += sp[(n % V_) * D_ + d] + tp[(n % TT_) * D_ + d];
  h[i] = acc;
}

// ---------------------------------------------------------------------------
// Sinusoidal timestep embedding + 2-layer SiLU MLP. One block of 256 threads.
// ---------------------------------------------------------------------------
__global__ void time_embed_kernel(const float* __restrict__ t,
                                  const float* __restrict__ w1,
                                  const float* __restrict__ b1,
                                  const float* __restrict__ w2,
                                  const float* __restrict__ b2,
                                  float* __restrict__ te,
                                  float* __restrict__ tes) {
  __shared__ float raw[D_];
  __shared__ float h1[D_];
  int tid = threadIdx.x;
  for (int b = 0; b < B_; ++b) {
    float tv = t[b];
    float fr, v;
    if (tid < 128) { fr = __expf(-logf(10000.0f) * tid / 128.0f);        v = cosf(tv * fr); }
    else           { fr = __expf(-logf(10000.0f) * (tid - 128) / 128.0f); v = sinf(tv * fr); }
    raw[tid] = v;
    __syncthreads();
    float a = b1[tid];
    for (int k = 0; k < D_; ++k) a += raw[k] * w1[k * D_ + tid];
    h1[tid] = silu_f(a);
    __syncthreads();
    float o = b2[tid];
    for (int k = 0; k < D_; ++k) o += h1[k] * w2[k * D_ + tid];
    te [b * D_ + tid] = o;
    tes[b * D_ + tid] = silu_f(o);
    __syncthreads();
  }
}

// ---------------------------------------------------------------------------
// All AdaLN modulations for all layers: mod{1,2}[l,b,j] = tes[b]·ada_w[l,:,j]+b
// ---------------------------------------------------------------------------
__global__ void adamod_kernel(const float* __restrict__ tes,
                              const float* __restrict__ a1w, const float* __restrict__ a1b,
                              const float* __restrict__ a2w, const float* __restrict__ a2b,
                              float* __restrict__ mod1, float* __restrict__ mod2) {
  int per = L_ * B_ * 2 * D_;
  int i = blockIdx.x * blockDim.x + threadIdx.x;
  if (i >= 2 * per) return;
  int set = i / per, r = i - set * per;
  int l = r / (B_ * 2 * D_);
  int rem = r - l * (B_ * 2 * D_);
  int b = rem / (2 * D_);
  int j = rem - b * (2 * D_);
  const float* w  = set ? a2w : a1w;
  const float* bi = set ? a2b : a1b;
  float acc = bi[l * 2 * D_ + j];
  const float* tv = tes + b * D_;
  const float* wp = w + (size_t)l * D_ * 2 * D_ + j;
  for (int k = 0; k < D_; ++k) acc += tv[k] * wp[(size_t)k * 2 * D_];
  (set ? mod2 : mod1)[r] = acc;
}

// ---------------------------------------------------------------------------
// LayerNorm + AdaLN modulate, f16 output. One row (D=256) per 256-thread block.
// ---------------------------------------------------------------------------
__global__ void ln_mod_kernel(const float* __restrict__ h,
                              const float* __restrict__ mod,
                              _Float16* __restrict__ out16) {
  int row = blockIdx.x;            // 0..B*N-1
  int tid = threadIdx.x;
  int b = row / N_;
  __shared__ float sdata[D_];
  __shared__ float stat[2];
  float x = h[(size_t)row * D_ + tid];
  sdata[tid] = x; __syncthreads();
  for (int s = 128; s > 0; s >>= 1) { if (tid < s) sdata[tid] += sdata[tid + s]; __syncthreads(); }
  if (tid == 0) stat[0] = sdata[0] * (1.0f / D_);
  __syncthreads();
  float m = stat[0], d = x - m;
  sdata[tid] = d * d; __syncthreads();
  for (int s = 128; s > 0; s >>= 1) { if (tid < s) sdata[tid] += sdata[tid + s]; __syncthreads(); }
  if (tid == 0) stat[1] = rsqrtf(sdata[0] * (1.0f / D_) + 1e-5f);
  __syncthreads();
  float v  = d * stat[1];
  float sh = mod[b * 2 * D_ + tid];
  float sc = mod[b * 2 * D_ + D_ + tid];
  out16[(size_t)row * D_ + tid] = (_Float16)(v * (1.0f + sc) + sh);
}

// Final LN with affine gamma/beta, in-place fp32.
__global__ void ln_final_kernel(float* h, const float* __restrict__ g,
                                const float* __restrict__ bb) {
  int row = blockIdx.x;
  int tid = threadIdx.x;
  __shared__ float sdata[D_];
  __shared__ float stat[2];
  float x = h[(size_t)row * D_ + tid];
  sdata[tid] = x; __syncthreads();
  for (int s = 128; s > 0; s >>= 1) { if (tid < s) sdata[tid] += sdata[tid + s]; __syncthreads(); }
  if (tid == 0) stat[0] = sdata[0] * (1.0f / D_);
  __syncthreads();
  float m = stat[0], d = x - m;
  sdata[tid] = d * d; __syncthreads();
  for (int s = 128; s > 0; s >>= 1) { if (tid < s) sdata[tid] += sdata[tid + s]; __syncthreads(); }
  if (tid == 0) stat[1] = rsqrtf(sdata[0] * (1.0f / D_) + 1e-5f);
  __syncthreads();
  h[(size_t)row * D_ + tid] = d * stat[1] * g[tid] + bb[tid];
}

// ---------------------------------------------------------------------------
// Pack fp32 QKV (B*N,3D) into f16 q/k as (bh, n, 32) and V transposed (bh, 32, n)
// so WMMA B-fragments (K-contiguous columns) come straight from memory.
// ---------------------------------------------------------------------------
__global__ void qkv_pack_kernel(const float* __restrict__ qkv,
                                _Float16* __restrict__ q16,
                                _Float16* __restrict__ k16,
                                _Float16* __restrict__ vT) {
  size_t i = (size_t)blockIdx.x * blockDim.x + threadIdx.x;
  size_t tot = (size_t)B_ * H_ * N_ * DH_;
  if (i >= tot) return;
  int e  = (int)(i % DH_);
  int n  = (int)((i / DH_) % N_);
  int hh = (int)((i / ((size_t)DH_ * N_)) % H_);
  int b  = (int)(i / ((size_t)DH_ * N_ * H_));
  int bh = b * H_ + hh;
  const float* base = qkv + ((size_t)b * N_ + n) * (3 * D_);
  q16[((size_t)bh * N_ + n) * DH_ + e] = (_Float16)base[hh * DH_ + e];
  k16[((size_t)bh * N_ + n) * DH_ + e] = (_Float16)base[D_ + hh * DH_ + e];
  vT [((size_t)bh * DH_ + e) * N_ + n] = (_Float16)base[2 * D_ + hh * DH_ + e];
}

// ---------------------------------------------------------------------------
// Attention: one workgroup per (q-block-of-16, b*h). 4 waves / 128 threads.
// Score strip 16x2400 lives in LDS (153.6 KB of the 320 KB/WGP budget).
//  Phase 1: S = (Q Kᵀ)*scale + bias   (one WMMA per 16x16 tile, d_head=32=K)
//  Phase 2: row softmax in LDS
//  Phase 3: O = P V via WMMA, K split across waves, LDS reduction
// ---------------------------------------------------------------------------
__global__ void attn_kernel(const _Float16* __restrict__ q16,
                            const _Float16* __restrict__ k16,
                            const _Float16* __restrict__ vT,
                            const float* __restrict__ bias,
                            _Float16* __restrict__ o16) {
  extern __shared__ float smem[];
  float* S       = smem;                 // 16*N_
  float* red     = smem + 16 * N_;       // 128
  float* rowstat = red + 128;            // 16
  float* oacc    = rowstat + 16;         // 4*512

  int qb = blockIdx.x, bh = blockIdx.y;
  int wave = threadIdx.x >> 5, lane = threadIdx.x & 31;
  const float scale = 0.17677669529663689f;   // 1/sqrt(32)

  const _Float16* qbase = q16 + ((size_t)bh * N_ + qb * 16) * DH_;
  const _Float16* kbase = k16 + (size_t)bh * N_ * DH_;
  const _Float16* vbase = vT  + (size_t)bh * DH_ * N_;

  v16h qf = load_frag16(qbase, DH_, 0, 0, lane);
  int mrow = (lane >> 4) << 3;
  int cn   = lane & 15;

  for (int j = wave; j < N_ / 16; j += 4) {
    v16h kf = load_frag16(kbase, DH_, j * 16, 0, lane);
    v8f s = {};
    s = __builtin_amdgcn_wmma_f32_16x16x32_f16(false, qf, false, kf,
                                               (short)0, s, false, false);
    int col = j * 16 + cn;
#pragma unroll
    for (int r = 0; r < 8; ++r) {
      int m = mrow + r;
      S[m * N_ + col] = s[r] * scale + bias[(size_t)(qb * 16 + m) * N_ + col];
    }
  }
  __syncthreads();

  // --- softmax: 8 threads per row ---
  int t = threadIdx.x, row = t >> 3, off = t & 7;
  float mx = -3.0e38f;
  for (int c = off; c < N_; c += 8) mx = fmaxf(mx, S[row * N_ + c]);
  red[t] = mx; __syncthreads();
  if (t < 16) { float m2 = red[t * 8];
    for (int i2 = 1; i2 < 8; ++i2) m2 = fmaxf(m2, red[t * 8 + i2]);
    rowstat[t] = m2; }
  __syncthreads();
  float m = rowstat[row];
  float sum = 0.0f;
  for (int c = off; c < N_; c += 8) {
    float e = __expf(S[row * N_ + c] - m);
    S[row * N_ + c] = e; sum += e;
  }
  red[t] = sum; __syncthreads();
  if (t < 16) { float s2 = 0.0f;
    for (int i2 = 0; i2 < 8; ++i2) s2 += red[t * 8 + i2];
    rowstat[t] = s2; }
  __syncthreads();
  float inv = 1.0f / rowstat[row];
  for (int c = off; c < N_; c += 8) S[row * N_ + c] *= inv;
  __syncthreads();

  // --- O = P @ V : two 16-wide column tiles (d_head=32), K split over waves ---
  v8f a0 = {}, a1 = {};
  int mA = lane & 15, koffA = (lane >> 4) << 3;
  for (int j = wave; j < N_ / 32; j += 4) {
    int k0 = j * 32;
    const float* p = &S[mA * N_ + k0 + koffA];
    v16h pf;
#pragma unroll
    for (int i2 = 0; i2 < 8; ++i2) { pf[i2] = (_Float16)p[i2]; pf[i2 + 8] = (_Float16)p[16 + i2]; }
    v16h v0 = load_frag16(vbase, N_, 0,  k0, lane);
    v16h v1 = load_frag16(vbase, N_, 16, k0, lane);
    a0 = __builtin_amdgcn_wmma_f32_16x16x32_f16(false, pf, false, v0, (short)0, a0, false, false);
    a1 = __builtin_amdgcn_wmma_f32_16x16x32_f16(false, pf, false, v1, (short)0, a1, false, false);
  }
#pragma unroll
  for (int r = 0; r < 8; ++r) {
    oacc[wave * 512 + (mrow + r) * 32 + cn]      = a0[r];
    oacc[wave * 512 + (mrow + r) * 32 + 16 + cn] = a1[r];
  }
  __syncthreads();
  int b = bh >> 3, hh = bh & 7;
  for (int idx = threadIdx.x; idx < 512; idx += 128) {
    float v2 = oacc[idx] + oacc[512 + idx] + oacc[1024 + idx] + oacc[1536 + idx];
    int m3 = idx >> 5, e = idx & 31;
    o16[((size_t)b * N_ + qb * 16 + m3) * D_ + hh * DH_ + e] = (_Float16)v2;
  }
}

// ---------------------------------------------------------------------------
// ht[b,v,d,t] = sum_tt h[b, v*TT+tt, d] * tproj_w[tt,t] + tproj_b[t]
// ---------------------------------------------------------------------------
__global__ void tproj_kernel(const float* __restrict__ h,
                             const float* __restrict__ w,
                             const float* __restrict__ bias,
                             float* __restrict__ ht) {
  size_t i = (size_t)blockIdx.x * blockDim.x + threadIdx.x;
  if (i >= (size_t)B_ * V_ * D_ * T_) return;
  int t = (int)(i % T_);
  int d = (int)((i / T_) % D_);
  int v = (int)((i / ((size_t)T_ * D_)) % V_);
  int b = (int)(i / ((size_t)T_ * D_ * V_));
  float acc = bias[t];
  for (int tt = 0; tt < TT_; ++tt)
    acc += h[((size_t)b * N_ + v * TT_ + tt) * D_ + d] * w[tt * T_ + t];
  ht[i] = acc;
}

// Pack ht (b,v,d,t) into row-major f16 rows (b*V+v)*T+t over d for WMMA.
__global__ void packg_kernel(const float* __restrict__ ht,
                             _Float16* __restrict__ g16) {
  size_t i = (size_t)blockIdx.x * blockDim.x + threadIdx.x;
  if (i >= (size_t)B_ * V_ * T_ * D_) return;
  int d = (int)(i % D_);
  size_t row = i / D_;
  int t = (int)(row % T_);
  size_t bv = row / T_;
  g16[i] = (_Float16)ht[(bv * D_ + d) * T_ + t];
}

// out[b,f,v,t] = buf2[row]·out_w2[:,f] + b2[f]
__global__ void outfinal_kernel(const _Float16* __restrict__ buf2,
                                const float* __restrict__ w2,
                                const float* __restrict__ b2,
                                float* __restrict__ out) {
  int i = blockIdx.x * blockDim.x + threadIdx.x;
  if (i >= B_ * V_ * T_ * FEAT_) return;
  int f = i % FEAT_;
  int row = i / FEAT_;
  int t = row % T_;
  int bv = row / T_;
  int v = bv % V_, b = bv / V_;
  float acc = b2[f];
  const _Float16* pr = buf2 + (size_t)row * D_;
  for (int e = 0; e < D_; ++e) acc += (float)pr[e] * w2[e * FEAT_ + f];
  out[(((size_t)b * FEAT_ + f) * V_ + v) * T_ + t] = acc;
}

// ===========================================================================
extern "C" void kernel_launch(void* const* d_in, const int* in_sizes, int n_in,
                              void* d_out, int out_size, void* d_ws, size_t ws_size,
                              hipStream_t stream) {
  (void)in_sizes; (void)n_in; (void)out_size; (void)ws_size;

  const float* x        = (const float*)d_in[0];
  const float* x_masked = (const float*)d_in[1];
  const float* tvec     = (const float*)d_in[2];
  const float* attnbias = (const float*)d_in[3];
  const float* xproj_w  = (const float*)d_in[4];
  const float* xproj_b  = (const float*)d_in[5];
  const float* tmlp_w1  = (const float*)d_in[6];
  const float* tmlp_b1  = (const float*)d_in[7];
  const float* tmlp_w2  = (const float*)d_in[8];
  const float* tmlp_b2  = (const float*)d_in[9];
  const float* sp_emb   = (const float*)d_in[10];
  const float* tp_emb   = (const float*)d_in[11];
  const float* ada1_w   = (const float*)d_in[12];
  const float* ada1_b   = (const float*)d_in[13];
  const float* wqkv     = (const float*)d_in[14];
  const float* bqkv     = (const float*)d_in[15];
  const float* wo       = (const float*)d_in[16];
  const float* bo       = (const float*)d_in[17];
  const float* ada2_w   = (const float*)d_in[18];
  const float* ada2_b   = (const float*)d_in[19];
  const float* ffn_w1   = (const float*)d_in[20];
  const float* ffn_b1   = (const float*)d_in[21];
  const float* ffn_w2   = (const float*)d_in[22];
  const float* ffn_b2   = (const float*)d_in[23];
  const float* fn_g     = (const float*)d_in[24];
  const float* fn_b     = (const float*)d_in[25];
  const float* tproj_w  = (const float*)d_in[26];
  const float* tproj_b  = (const float*)d_in[27];
  const float* out_w1   = (const float*)d_in[28];
  const float* out_b1   = (const float*)d_in[29];
  const float* out_w2   = (const float*)d_in[30];
  const float* out_b2   = (const float*)d_in[31];

  // ---- workspace carve (256B aligned) ----
  char* p = (char*)d_ws;
  auto alloc = [&](size_t bytes) -> void* {
    void* r = (void*)p;
    p += (bytes + 255) & ~(size_t)255;
    return r;
  };
  const int BN = B_ * N_;                       // 4800
  float* h    = (float*)alloc((size_t)BN * D_ * 4);
  float* te   = (float*)alloc((size_t)B_ * D_ * 4);
  float* tes  = (float*)alloc((size_t)B_ * D_ * 4);
  float* mod1 = (float*)alloc((size_t)L_ * B_ * 2 * D_ * 4);
  float* mod2 = (float*)alloc((size_t)L_ * B_ * 2 * D_ * 4);
  float* qkv  = (float*)alloc((size_t)BN * 3 * D_ * 4);
  float* ht   = (float*)alloc((size_t)B_ * V_ * D_ * T_ * 4);
  _Float16* xm16  = (_Float16*)alloc((size_t)BN * D_ * 2);
  _Float16* o16   = (_Float16*)alloc((size_t)BN * D_ * 2);
  _Float16* mid16 = (_Float16*)alloc((size_t)BN * 4 * D_ * 2);
  _Float16* g16   = (_Float16*)alloc((size_t)B_ * V_ * T_ * D_ * 2);
  _Float16* buf2  = (_Float16*)alloc((size_t)B_ * V_ * T_ * D_ * 2);
  _Float16* q16   = (_Float16*)alloc((size_t)B_ * H_ * N_ * DH_ * 2);
  _Float16* k16   = (_Float16*)alloc((size_t)B_ * H_ * N_ * DH_ * 2);
  _Float16* vTd   = (_Float16*)alloc((size_t)B_ * H_ * N_ * DH_ * 2);
  _Float16* wqkvT = (_Float16*)alloc((size_t)L_ * 3 * D_ * D_ * 2);
  _Float16* woT   = (_Float16*)alloc((size_t)L_ * D_ * D_ * 2);
  _Float16* f1T   = (_Float16*)alloc((size_t)L_ * 4 * D_ * D_ * 2);
  _Float16* f2T   = (_Float16*)alloc((size_t)L_ * 4 * D_ * D_ * 2);
  _Float16* ow1T  = (_Float16*)alloc((size_t)D_ * D_ * 2);

  auto cdiv = [](size_t a, size_t b) { return (int)((a + b - 1) / b); };

  // ---- weight transposes (fp32 -> f16, K-contiguous for WMMA B fragments) --
  wtrans_kernel<<<cdiv((size_t)L_ * D_ * 3 * D_, 256), 256, 0, stream>>>(wqkv, wqkvT, D_, 3 * D_, L_);
  wtrans_kernel<<<cdiv((size_t)L_ * D_ * D_,     256), 256, 0, stream>>>(wo, woT, D_, D_, L_);
  wtrans_kernel<<<cdiv((size_t)L_ * D_ * 4 * D_, 256), 256, 0, stream>>>(ffn_w1, f1T, D_, 4 * D_, L_);
  wtrans_kernel<<<cdiv((size_t)L_ * 4 * D_ * D_, 256), 256, 0, stream>>>(ffn_w2, f2T, 4 * D_, D_, L_);
  wtrans_kernel<<<cdiv((size_t)D_ * D_,          256), 256, 0, stream>>>(out_w1, ow1T, D_, D_, 1);

  // ---- embedding + positional ----
  embed_kernel<<<cdiv((size_t)BN * D_, 256), 256, 0, stream>>>(
      x, x_masked, xproj_w, xproj_b, sp_emb, tp_emb, h);

  // ---- timestep embedding + AdaLN modulations (all layers) ----
  time_embed_kernel<<<1, 256, 0, stream>>>(tvec, tmlp_w1, tmlp_b1, tmlp_w2, tmlp_b2, te, tes);
  adamod_kernel<<<cdiv((size_t)2 * L_ * B_ * 2 * D_, 256), 256, 0, stream>>>(
      tes, ada1_w, ada1_b, ada2_w, ada2_b, mod1, mod2);

  // Block = 8 waves sharing one async-staged A panel; LDS = 16*K halves.
  auto gemm = [&](const _Float16* A, const _Float16* WT, const float* bias,
                  const float* resid, float* outf, _Float16* outh,
                  int M, int Nc, int K, int g) {
    int tnGroups = (Nc / 16) / 8;               // all Nc here are multiples of 128
    int grid = (M / 16) * tnGroups;
    gemm_wmma_f16<<<grid, 256, 16 * K * 2, stream>>>(A, WT, bias, resid,
                                                     outf, outh, Nc, K, g,
                                                     tnGroups);
  };

  const int SHBYTES = (16 * N_ + 128 + 16 + 4 * 512) * (int)sizeof(float); // 162,368 B

  for (int l = 0; l < L_; ++l) {
    // AdaLN 1 + QKV projection
    ln_mod_kernel<<<BN, 256, 0, stream>>>(h, mod1 + (size_t)l * B_ * 2 * D_, xm16);
    gemm(xm16, wqkvT + (size_t)l * 3 * D_ * D_, bqkv + l * 3 * D_,
         nullptr, qkv, nullptr, BN, 3 * D_, D_, 0);
    qkv_pack_kernel<<<cdiv((size_t)B_ * H_ * N_ * DH_, 256), 256, 0, stream>>>(qkv, q16, k16, vTd);
    // masked MHA (LDS-resident score strips)
    attn_kernel<<<dim3(N_ / 16, B_ * H_), 128, SHBYTES, stream>>>(q16, k16, vTd, attnbias, o16);
    // output projection + residual
    gemm(o16, woT + (size_t)l * D_ * D_, bo + l * D_, h, h, nullptr, BN, D_, D_, 0);
    // AdaLN 2 + FFN
    ln_mod_kernel<<<BN, 256, 0, stream>>>(h, mod2 + (size_t)l * B_ * 2 * D_, xm16);
    gemm(xm16, f1T + (size_t)l * 4 * D_ * D_, ffn_b1 + l * 4 * D_,
         nullptr, nullptr, mid16, BN, 4 * D_, D_, 1 /*gelu*/);
    gemm(mid16, f2T + (size_t)l * 4 * D_ * D_, ffn_b2 + l * D_,
         h, h, nullptr, BN, D_, 4 * D_, 0);
  }

  // ---- head ----
  ln_final_kernel<<<BN, 256, 0, stream>>>(h, fn_g, fn_b);
  tproj_kernel<<<cdiv((size_t)B_ * V_ * D_ * T_, 256), 256, 0, stream>>>(h, tproj_w, tproj_b, ht);
  packg_kernel<<<cdiv((size_t)B_ * V_ * T_ * D_, 256), 256, 0, stream>>>(ht, g16);
  gemm(g16, ow1T, out_b1, nullptr, nullptr, buf2, B_ * V_ * T_, D_, D_, 1 /*gelu*/);
  outfinal_kernel<<<cdiv((size_t)B_ * V_ * T_ * FEAT_, 256), 256, 0, stream>>>(
      buf2, out_w2, out_b2, (float*)d_out);
}
